// AdaptiveFNO2d_15006615734086
// MI455X (gfx1250) — compile-verified, matrix operand order checked by reading, not run on hardware
//
#include <hip/hip_runtime.h>
#include <hip/hip_bf16.h>
#include <math.h>

// ---------------------------------------------------------------------------
// AdaptiveFNO2d on gfx1250 (MI455X): all DFTs + channel matmuls as f16 WMMA
// (v_wmma_f32_16x16x32_f16), f32 accumulation. All matrices stored so the
// WMMA K axis is memory-contiguous (vector b128 tile loads), and C tiles are
// oriented so M lands on the output's contiguous axis (vector b128 stores).
// ---------------------------------------------------------------------------

#define NB   128      // batch
#define NC   64       // width / channels
#define NH   64       // H (and kh modes)
#define NW   126      // W
#define WP   128      // padded W pitch
#define NK   64       // rfft modes along W (W/2+1)
#define NL   4        // layers

typedef __attribute__((ext_vector_type(16))) _Float16 v16h;
typedef __attribute__((ext_vector_type(8)))  _Float16 v8h;
typedef __attribute__((ext_vector_type(8)))  float    v8f;
typedef __attribute__((ext_vector_type(4)))  float    f4;

#define DEV __device__ __forceinline__

// ----- WMMA core (wave32, 16x16 C tile, K=32 per step) ---------------------
DEV v8f wmma_f16(v16h a, v16h b, v8f c) {
#if defined(__HIP_DEVICE_COMPILE__)
  return __builtin_amdgcn_wmma_f32_16x16x32_f16(
      /*neg_a=*/false, a, /*neg_b=*/false, b,
      /*c_mod=*/(short)0, c, /*reuse_a=*/false, /*reuse_b=*/false);
#else
  (void)a; (void)b; return c;
#endif
}

// Per-lane WMMA 16x32 tile layout (ISA 7.12.2): lane owns row/col (lane&15),
// K = {kb..kb+7, kb+16..kb+23} with kb = (lane>>4)*8 -> two contiguous runs.

// f16 tile, K contiguous: two b128 loads.
DEV v16h ld_h16v(const _Float16* base, long aS, int lane) {
  const _Float16* p = base + (long)(lane & 15) * aS + (long)((lane >> 4) << 3);
  v8h lo = *(const v8h*)p;
  v8h hi = *(const v8h*)(p + 16);
  v16h r;
#pragma unroll
  for (int j = 0; j < 8; ++j) { r[j] = lo[j]; r[j + 8] = hi[j]; }
  return r;
}
// f32 tile, K contiguous: four b128 loads + pack-convert.
DEV v16h ld_f32v(const float* base, long aS, int lane) {
  const float* p = base + (long)(lane & 15) * aS + (long)((lane >> 4) << 3);
  f4 a0 = *(const f4*)p,        a1 = *(const f4*)(p + 4);
  f4 b0 = *(const f4*)(p + 16), b1 = *(const f4*)(p + 20);
  v16h r;
#pragma unroll
  for (int j = 0; j < 4; ++j) {
    r[j]      = (_Float16)a0[j];
    r[j + 4]  = (_Float16)a1[j];
    r[j + 8]  = (_Float16)b0[j];
    r[j + 12] = (_Float16)b1[j];
  }
  return r;
}
// f32 tile, K strided (lanes still coalesce along aS=1 axis).
DEV v16h ld_f32s(const float* base, long aS, long kS, int lane) {
  const float* p = base + (long)(lane & 15) * aS + (long)((lane >> 4) << 3) * kS;
  v16h r;
#pragma unroll
  for (int j = 0; j < 8; ++j) {
    r[j]     = (_Float16)p[(long)j * kS];
    r[j + 8] = (_Float16)p[(long)(j + 16) * kS];
  }
  return r;
}
DEV void st_f8(float* p, const v8f& a) {
  f4 lo = {a[0], a[1], a[2], a[3]};
  f4 hi = {a[4], a[5], a[6], a[7]};
  *(f4*)p = lo;
  *(f4*)(p + 4) = hi;
}

// ----- init: DFT matrices (f16) + sigmoid(mode_weights) --------------------
__global__ void k_init(const float* __restrict__ modew, float* __restrict__ mw,
                       _Float16* FwcT, _Float16* FwsT,   // [kw=64][w=128] fwd W (cols>=126 zero)
                       _Float16* Fhc, _Float16* Fhs,     // [kh=64][h=64]  fwd H (symmetric)
                       _Float16* iFhc, _Float16* iFhs,   // [h=64][kh=64]  inv H (1/64)
                       _Float16* iFwTc, _Float16* iFwTs) { // [w=128][kw=64] inv W (irfft fold)
  int id = blockIdx.x * 256 + threadIdx.x;               // 8192 threads
  const float TP = 6.283185307179586f;
  if (id < 64 * 128) {                                   // FwT[k][w] : e^{-i 2pi w k/126}
    int k = id >> 7, w = id & 127;
    float th = TP * (float)(w * k) / 126.f;
    FwcT[id] = (_Float16)((w < NW) ?  cosf(th) : 0.f);
    FwsT[id] = (_Float16)((w < NW) ? -sinf(th) : 0.f);
  }
  if (id < 64 * 64) {
    int a = id >> 6, h = id & 63;
    float th = TP * (float)(a * h) / 64.f;
    Fhc[id]  = (_Float16)cosf(th);
    Fhs[id]  = (_Float16)sinf(th);
    iFhc[id] = (_Float16)(cosf(th) * (1.f / 64.f));
    iFhs[id] = (_Float16)(sinf(th) * (1.f / 64.f));
    mw[id] = 1.f / (1.f + expf(-modew[id]));
  }
  if (id < 128 * 64) {                                   // iFwT[w][k], Hermitian fold
    int w = id >> 6, k = id & 63;
    float ck = ((k == 0) || (k == 63)) ? (1.f / 126.f) : (2.f / 126.f);
    float th = TP * (float)(k * w) / 126.f;
    iFwTc[id] = (_Float16)((w < NW) ? ( ck * cosf(th)) : 0.f);
    iFwTs[id] = (_Float16)((w < NW) ? (-ck * sinf(th)) : 0.f);
  }
}

// ----- pre-transpose spec_w: [l,i,o,kh,kw]f32 -> Wt[l,kh,kw,o,i]f16 --------
__global__ void k_twt(const float* __restrict__ sw, _Float16* __restrict__ Wt) {
  long id = (long)blockIdx.x * 256 + threadIdx.x;        // 67,108,864 total
  if (id >= (long)NL * 64 * 64 * 64 * 64) return;
  int i  = (int)(id & 63), o = (int)((id >> 6) & 63);
  int kw = (int)((id >> 12) & 63), kh = (int)((id >> 18) & 63);
  int l  = (int)(id >> 24);
  Wt[id] = (_Float16)sw[(((long)(l * 64 + i) * 64 + o) << 12) + kh * 64 + kw];
}

// ----- encode: 1x1 conv 3->64, into padded spatial [B,C,H,WP] --------------
__global__ void k_encode(const float* __restrict__ xin, const float* __restrict__ ew,
                         const float* __restrict__ eb, float* __restrict__ x) {
  long id = (long)blockIdx.x * 256 + threadIdx.x;        // B*64*64*128
  if (id >= (long)NB * NC * NH * WP) return;
  int w = (int)(id & 127), h = (int)((id >> 7) & 63);
  int o = (int)((id >> 13) & 63), b = (int)(id >> 19);
  float v = 0.f;
  if (w < NW) {
    const float* px = xin + ((long)(b * 3) * NH + h) * NW + w;
    v = eb[o];
#pragma unroll
    for (int c = 0; c < 3; ++c) v += ew[o * 3 + c] * px[(long)c * NH * NW];
  }
  x[id] = v;
}

// ----- forward W DFT: x[B,C,H,WP] -> t[B,H,KW,C] (re,im) -------------------
// rows m = b*4096 + h*64 + c (c fastest) -> contiguous stores over c.
__global__ void k_fwdW(const float* __restrict__ x, const _Float16* __restrict__ FwcT,
                       const _Float16* __restrict__ FwsT,
                       float* __restrict__ tre, float* __restrict__ tim) {
  int wid  = (blockIdx.x << 3) + (threadIdx.x >> 5);     // 131072 waves
  int lane = threadIdx.x & 31;
  int kwT = wid & 3, rT = wid >> 2;
  int m0 = rT << 4;
  int b = m0 >> 12, h = (m0 >> 6) & 63, c0 = m0 & 63;
  const float* A = x + (long)b * 524288 + (long)c0 * 8192 + (long)h * 128;
  v8f aR = {}, aI = {};
#pragma unroll
  for (int ks = 0; ks < 4; ++ks) {                       // K = 128 (>=126, zero-padded)
    v16h a  = ld_f32v(A + ks * 32, 8192, lane);          // A[m=c][k=w]
    v16h br = ld_h16v(FwcT + (long)(kwT * 16) * 128 + ks * 32, 128, lane); // B[k=w][n=kw]
    v16h bi = ld_h16v(FwsT + (long)(kwT * 16) * 128 + ks * 32, 128, lane);
    aR = wmma_f16(a, br, aR);
    aI = wmma_f16(a, bi, aI);
  }
  int n = (kwT << 4) + (lane & 15);                      // kw
  int cb = c0 + ((lane >> 4) << 3);                      // c base (8 contiguous)
  long o = (long)b * 262144 + (long)h * 4096 + (long)n * 64 + cb;
  st_f8(tre + o, aR);
  st_f8(tim + o, aI);
}

// ----- H-direction complex DFT (fwd sigma=+1, inv sigma=-1), optional mw ---
// layout [B,(h|kh),KW,C]; per (b,kw): C[kh][c] = M[kh][h] * X[h][c]
__global__ void k_H(const float* __restrict__ ire, const float* __restrict__ iim,
                    const _Float16* __restrict__ Mc, const _Float16* __restrict__ Ms,
                    const float* __restrict__ mw, float sigma,
                    float* __restrict__ ore, float* __restrict__ oim) {
  int wid  = (blockIdx.x << 3) + (threadIdx.x >> 5);     // 131072 waves
  int lane = threadIdx.x & 31;
  int cT = wid & 3, khT = (wid >> 2) & 3, kw = (wid >> 4) & 63, b = wid >> 10;
  long bin = (long)b * 262144 + (long)kw * 64;
  __builtin_prefetch(ire + bin + (long)32 * 4096 + cT * 16 + (lane & 15), 0, 1);
  __builtin_prefetch(iim + bin + (long)32 * 4096 + cT * 16 + (lane & 15), 0, 1);
  v8f cr = {}, ci = {}, sr = {}, si = {};
#pragma unroll
  for (int ks = 0; ks < 2; ++ks) {                       // K = 64
    v16h ac = ld_h16v(Mc + (long)(khT * 16) * 64 + ks * 32, 64, lane);
    v16h as = ld_h16v(Ms + (long)(khT * 16) * 64 + ks * 32, 64, lane);
    v16h xr = ld_f32s(ire + bin + (long)ks * 32 * 4096 + cT * 16, 1, 4096, lane);
    v16h xi = ld_f32s(iim + bin + (long)ks * 32 * 4096 + cT * 16, 1, 4096, lane);
    cr = wmma_f16(ac, xr, cr); ci = wmma_f16(ac, xi, ci);
    sr = wmma_f16(as, xr, sr); si = wmma_f16(as, xi, si);
  }
  int n = cT * 16 + (lane & 15);
  int khb = khT * 16 + ((lane >> 4) << 3);
#pragma unroll
  for (int r = 0; r < 8; ++r) {
    int kh = khb + r;
    float re = cr[r] + sigma * si[r];
    float im = ci[r] - sigma * sr[r];
    float s = mw ? mw[kh * 64 + kw] : 1.f;
    long o = (long)b * 262144 + (long)kh * 4096 + (long)kw * 64 + n;
    ore[o] = re * s; oim[o] = im * s;
  }
}

// ----- spectral conv: per mode, C[o][b] = Wt[o][i] * X^T[i][b] (real wts) --
__global__ void k_spec(const float* __restrict__ xre, const float* __restrict__ xim,
                       const _Float16* __restrict__ Wt,
                       float* __restrict__ ore, float* __restrict__ oim) {
  int wid  = (blockIdx.x << 3) + (threadIdx.x >> 5);     // 131072 waves
  int lane = threadIdx.x & 31;
  int oT = wid & 3, bT = (wid >> 2) & 7, mode = wid >> 5;
  int kw = mode & 63, kh = mode >> 6;
  long ab = (long)kh * 4096 + (long)kw * 64;
  const _Float16* Am = Wt + (long)mode * 4096;           // [o][i], pitch 64
  v8f ar = {}, ai = {};
#pragma unroll
  for (int ks = 0; ks < 2; ++ks) {                       // K = 64 channels
    v16h a  = ld_h16v(Am + (long)(oT * 16) * 64 + ks * 32, 64, lane);       // A[m=o][k=i]
    v16h br = ld_f32v(xre + ab + (long)(bT * 16) * 262144 + ks * 32, 262144, lane); // B[k=i][n=b]
    v16h bi = ld_f32v(xim + ab + (long)(bT * 16) * 262144 + ks * 32, 262144, lane);
    ar = wmma_f16(a, br, ar);
    ai = wmma_f16(a, bi, ai);
  }
  int nb = bT * 16 + (lane & 15);                        // b
  int ob = oT * 16 + ((lane >> 4) << 3);                 // o base (contiguous)
  long o = (long)nb * 262144 + ab + ob;
  st_f8(ore + o, ar);
  st_f8(oim + o, ai);
}

// ----- inverse W (irfft fold): g[B,H,KW,C] -> spatial [B,C,H,WP] -----------
// C[w][c] = iFwT[w][kw] * g[kw][c] -> contiguous stores over w.
__global__ void k_invW(const float* __restrict__ gre, const float* __restrict__ gim,
                       const _Float16* __restrict__ iFwTc, const _Float16* __restrict__ iFwTs,
                       float* __restrict__ out) {
  int wid  = (blockIdx.x << 3) + (threadIdx.x >> 5);     // 262144 waves
  int lane = threadIdx.x & 31;
  int wT = wid & 7, cT = (wid >> 3) & 3, h = (wid >> 5) & 63, b = wid >> 11;
  long ab = (long)b * 262144 + (long)h * 4096;           // g element (kw,c): + kw*64 + c
  __builtin_prefetch(gre + ab + cT * 16 + (lane & 15), 0, 1);
  __builtin_prefetch(gim + ab + cT * 16 + (lane & 15), 0, 1);
  v8f acc = {};
#pragma unroll
  for (int ks = 0; ks < 2; ++ks) {                       // K = 64 modes
    v16h ac = ld_h16v(iFwTc + (long)(wT * 16) * 64 + ks * 32, 64, lane);  // A[m=w][k=kw]
    v16h as = ld_h16v(iFwTs + (long)(wT * 16) * 64 + ks * 32, 64, lane);
    v16h br = ld_f32s(gre + ab + cT * 16 + (long)ks * 32 * 64, 1, 64, lane); // B[k=kw][n=c]
    v16h bi = ld_f32s(gim + ab + cT * 16 + (long)ks * 32 * 64, 1, 64, lane);
    acc = wmma_f16(ac, br, acc);
    acc = wmma_f16(as, bi, acc);
  }
  int n = cT * 16 + (lane & 15);                         // c
  int wb = (wT << 4) + ((lane >> 4) << 3);               // w base (contiguous)
  st_f8(out + (long)b * 524288 + (long)n * 8192 + (long)h * 128 + wb, acc);
}

// ----- combine: x = gelu(mlp_w*xw + mlp_b + xsp + spec_b) ------------------
// C[p][o] = xw^T[p][i] * mlp_w^T[i][o] -> contiguous stores over p (=h*128+w).
__global__ void k_combine(const float* __restrict__ xw, const float* __restrict__ xsp,
                          const float* __restrict__ mw_, const float* __restrict__ mb_,
                          const float* __restrict__ sb_, float* __restrict__ xo) {
  int wid  = (blockIdx.x << 3) + (threadIdx.x >> 5);     // 262144 waves
  int lane = threadIdx.x & 31;
  int oT = wid & 3, pT = (wid >> 2) & 511, b = wid >> 11;
  long xb = (long)b * 524288;
  int m0 = pT << 4;
  __builtin_prefetch(xw + xb + m0 + (lane & 15), 0, 1);
  v8f acc = {};
#pragma unroll
  for (int ks = 0; ks < 2; ++ks) {                       // K = 64 channels
    v16h a = ld_f32s(xw + xb + m0 + (long)ks * 32 * 8192, 1, 8192, lane); // A[m=p][k=i]
    v16h w = ld_f32v(mw_ + (long)(oT * 16) * 64 + ks * 32, 64, lane);     // B[k=i][n=o]
    acc = wmma_f16(a, w, acc);
  }
  int n = (oT << 4) + (lane & 15);                       // o
  int pb = m0 + ((lane >> 4) << 3);                      // p base (contiguous)
  long adr = xb + (long)n * 8192 + pb;
  float bias = mb_[n] + sb_[n];
  f4 s0 = *(const f4*)(xsp + adr);
  f4 s1 = *(const f4*)(xsp + adr + 4);
  v8f v;
#pragma unroll
  for (int r = 0; r < 8; ++r) {
    float t = acc[r] + bias + (r < 4 ? s0[r] : s1[r - 4]);
    v[r] = 0.5f * t * (1.f + erff(t * 0.70710678118654752f));
  }
  st_f8(xo + adr, v);
}

// ----- decode: 1x1 conv 64->1 into d_out [B,1,H,126] -----------------------
__global__ void k_decode(const float* __restrict__ x, const float* __restrict__ dw,
                         const float* __restrict__ db, float* __restrict__ out) {
  long id = (long)blockIdx.x * 256 + threadIdx.x;        // B*64*126
  if (id >= (long)NB * NH * NW) return;
  int w = (int)(id % NW);
  int h = (int)((id / NW) % NH);
  int b = (int)(id / (NW * NH));
  const float* p = x + (long)b * 524288 + (long)h * 128 + w;
  float v = db[0];
#pragma unroll
  for (int c = 0; c < 64; ++c) v += dw[c] * p[(long)c * 8192];
  out[id] = v;
}

// ---------------------------------------------------------------------------
extern "C" void kernel_launch(void* const* d_in, const int* in_sizes, int n_in,
                              void* d_out, int out_size, void* d_ws, size_t ws_size,
                              hipStream_t stream) {
  (void)in_sizes; (void)n_in; (void)out_size; (void)ws_size;
  const float* x_in   = (const float*)d_in[0];
  const float* modew  = (const float*)d_in[1];
  const float* enc_w  = (const float*)d_in[2];
  const float* enc_b  = (const float*)d_in[3];
  const float* dec_w  = (const float*)d_in[4];
  const float* dec_b  = (const float*)d_in[5];
  const float* spec_w = (const float*)d_in[6];
  const float* spec_b = (const float*)d_in[7];
  const float* mlp_w  = (const float*)d_in[8];
  const float* mlp_b  = (const float*)d_in[9];

  char* wp = (char*)d_ws;
  auto alloc = [&](size_t bytes) -> void* {
    void* r = (void*)wp; wp += (bytes + 255) & ~(size_t)255; return r;
  };
  _Float16* FwcT  = (_Float16*)alloc(64 * 128 * 2);
  _Float16* FwsT  = (_Float16*)alloc(64 * 128 * 2);
  _Float16* Fhc   = (_Float16*)alloc(64 * 64 * 2);
  _Float16* Fhs   = (_Float16*)alloc(64 * 64 * 2);
  _Float16* iFhc  = (_Float16*)alloc(64 * 64 * 2);
  _Float16* iFhs  = (_Float16*)alloc(64 * 64 * 2);
  _Float16* iFwTc = (_Float16*)alloc(128 * 64 * 2);
  _Float16* iFwTs = (_Float16*)alloc(128 * 64 * 2);
  float*    mwb   = (float*)alloc(64 * 64 * 4);
  _Float16* Wt    = (_Float16*)alloc((size_t)NL * 64 * 64 * 64 * 64 * 2); // 134 MB
  const size_t SPATIAL = (size_t)NB * NC * NH * WP;                       // 67.1M f32
  const size_t SPEC    = (size_t)NB * NC * NH * NK;                       // 33.6M f32
  float* xb   = (float*)alloc(SPATIAL * 4);
  float* xwb  = (float*)alloc(SPATIAL * 4);
  float* xspb = (float*)alloc(SPATIAL * 4);
  float* cAre = (float*)alloc(SPEC * 4);
  float* cAim = (float*)alloc(SPEC * 4);
  float* cBre = (float*)alloc(SPEC * 4);
  float* cBim = (float*)alloc(SPEC * 4);

  k_init<<<32, 256, 0, stream>>>(modew, mwb, FwcT, FwsT, Fhc, Fhs, iFhc, iFhs, iFwTc, iFwTs);
  k_twt<<<262144, 256, 0, stream>>>(spec_w, Wt);
  k_encode<<<262144, 256, 0, stream>>>(x_in, enc_w, enc_b, xb);

  for (int l = 0; l < NL; ++l) {
    // forward rfft2 with fused mode weighting -> cB = sigma(mw) * rfft2(x)
    k_fwdW<<<16384, 256, 0, stream>>>(xb, FwcT, FwsT, cAre, cAim);
    k_H<<<16384, 256, 0, stream>>>(cAre, cAim, Fhc, Fhs, mwb, +1.f, cBre, cBim);
    // MLP branch spatial input: x_w = irfft2(cB)
    k_H<<<16384, 256, 0, stream>>>(cBre, cBim, iFhc, iFhs, nullptr, -1.f, cAre, cAim);
    k_invW<<<32768, 256, 0, stream>>>(cAre, cAim, iFwTc, iFwTs, xwb);
    // spectral branch: of = spec_w (x) cB   (rfft2(irfft2(cB)) == cB fusion)
    k_spec<<<16384, 256, 0, stream>>>(cBre, cBim, Wt + (size_t)l * 16777216, cAre, cAim);
    k_H<<<16384, 256, 0, stream>>>(cAre, cAim, iFhc, iFhs, nullptr, -1.f, cBre, cBim);
    k_invW<<<32768, 256, 0, stream>>>(cBre, cBim, iFwTc, iFwTs, xspb);
    // x = gelu(conv1x1(x_w) + mlp_b + x_sp + spec_b)
    k_combine<<<32768, 256, 0, stream>>>(xwb, xspb, mlp_w + l * 4096, mlp_b + l * 64,
                                         spec_b + l * 64, xb);
  }
  k_decode<<<4032, 256, 0, stream>>>(xb, dec_w, dec_b, (float*)d_out);
}
